// EncoderLayer_84447646974625
// MI455X (gfx1250) — compile-verified, matrix operand order checked by reading, not run on hardware
//
#include <hip/hip_runtime.h>

// ---------------------------------------------------------------------------
// CDNA5 (gfx1250) transformer encoder layer, BF16-WMMA everywhere.
// Double-buffered LDS staging via GLOBAL_LOAD_ASYNC_TO_LDS_B128.
// ---------------------------------------------------------------------------

#define DEV __device__ __forceinline__
#define AS1 __attribute__((address_space(1)))
#define AS3 __attribute__((address_space(3)))

typedef int v4i_vs __attribute__((vector_size(16)));  // matches builtin proto

#if defined(__has_builtin)
#if __has_builtin(__builtin_amdgcn_global_load_async_to_lds_b128)
#define HAVE_ASYNC 1
#endif
#endif
#ifndef HAVE_ASYNC
#define HAVE_ASYNC 0
#endif

#if HAVE_ASYNC
#if __has_builtin(__builtin_amdgcn_s_wait_asynccnt)
#define WAIT_ASYNC() __builtin_amdgcn_s_wait_asynccnt(0)
#else
#define WAIT_ASYNC() asm volatile("s_wait_asynccnt 0x0" ::: "memory")
#endif
#define ASYNC_B128(gsrc, ldst)                                            \
  __builtin_amdgcn_global_load_async_to_lds_b128(                         \
      (AS1 v4i_vs*)(unsigned long long)(const void*)(gsrc),               \
      (AS3 v4i_vs*)(ldst), 0, 0)
#else
#define WAIT_ASYNC()
#endif

typedef __attribute__((ext_vector_type(16))) __bf16 v16bf;
typedef __attribute__((ext_vector_type(8)))  float  v8f;

union FragBF { v16bf v; unsigned int u[8]; };

DEV unsigned short f32_to_bf16(float f) {
  unsigned int u = __float_as_uint(f);
  u += 0x7FFFu + ((u >> 16) & 1u);     // round-to-nearest-even
  return (unsigned short)(u >> 16);
}

// ---------------------------------------------------------------------------
// fp32 -> bf16 elementwise conversion
// ---------------------------------------------------------------------------
__global__ __launch_bounds__(256)
void cvt_f32_to_bf16(const float* __restrict__ src,
                     unsigned short* __restrict__ dst, int n) {
  int i = blockIdx.x * blockDim.x + threadIdx.x;
  int stride = gridDim.x * blockDim.x;
  for (; i < n; i += stride) dst[i] = f32_to_bf16(src[i]);
}

// ---------------------------------------------------------------------------
// Tile staging: global -> LDS (async b128 where the data is already bf16).
// ---------------------------------------------------------------------------
#define BM 128
#define BN 128
#define BK 32
#define APITCH 40   // bf16 elems/row: 80B pitch -> 16B-aligned, bank-friendly
#define BPITCH 40

template <bool A_IS_F32, bool TRANS_B>
DEV void stage_tiles(unsigned short* AsT, unsigned short* BsT,
                     const float* __restrict__ Af,
                     const unsigned short* __restrict__ Ah,
                     const unsigned short* __restrict__ Bh,
                     int blockM, int blockN, int k0,
                     int lda, int ldb, int N, int K, int tid) {
  // ---- A tile (AsT[m][k]) : 128 rows x 64B = 2 b128 per thread ------------
  if (A_IS_F32) {
#pragma unroll
    for (int it = 0; it < 2; ++it) {
      const int idx = tid + (it << 8);
      const int m = idx >> 2;           // 0..127
      const int c = idx & 3;            // 8-element k chunk
      const float* ap = Af + (size_t)(blockM + m) * lda + (k0 + (c << 3));
      const float4 f0 = *(const float4*)ap;
      const float4 f1 = *(const float4*)(ap + 4);
      uint4 pk;
      pk.x = (unsigned)f32_to_bf16(f0.x) | ((unsigned)f32_to_bf16(f0.y) << 16);
      pk.y = (unsigned)f32_to_bf16(f0.z) | ((unsigned)f32_to_bf16(f0.w) << 16);
      pk.z = (unsigned)f32_to_bf16(f1.x) | ((unsigned)f32_to_bf16(f1.y) << 16);
      pk.w = (unsigned)f32_to_bf16(f1.z) | ((unsigned)f32_to_bf16(f1.w) << 16);
      *(uint4*)&AsT[m * APITCH + (c << 3)] = pk;
    }
    if (k0 + BK < K)  // pull the tile after next toward L2/WGP$
      __builtin_prefetch(Af + (size_t)(blockM + (tid >> 1)) * lda + (k0 + BK), 0, 0);
  } else {
#pragma unroll
    for (int it = 0; it < 2; ++it) {
      const int idx = tid + (it << 8);
      const int m = idx >> 2;
      const int c = idx & 3;
      const unsigned short* g = Ah + (size_t)(blockM + m) * lda + (k0 + (c << 3));
#if HAVE_ASYNC
      ASYNC_B128(g, &AsT[m * APITCH + (c << 3)]);
#else
      *(uint4*)&AsT[m * APITCH + (c << 3)] = *(const uint4*)g;
#endif
    }
  }
  // ---- B tile (BsT[n][k] column-major) ------------------------------------
  if (TRANS_B) {
#pragma unroll
    for (int it = 0; it < 2; ++it) {
      const int idx = tid + (it << 8);
      const int n = idx >> 2;
      const int c = idx & 3;
      int gn = blockN + n;
      gn = gn < N ? gn : N - 1;         // branchless clamp
      const unsigned short* g = Bh + (size_t)gn * ldb + (k0 + (c << 3));
#if HAVE_ASYNC
      ASYNC_B128(g, &BsT[n * BPITCH + (c << 3)]);
#else
      *(uint4*)&BsT[n * BPITCH + (c << 3)] = *(const uint4*)g;
#endif
    }
  } else {
    // B given as [K,N]: b128 row loads + 16-bit LDS scatter transpose.
#pragma unroll
    for (int it = 0; it < 2; ++it) {
      const int idx = tid + (it << 8);
      const int k = idx >> 4;           // 0..31
      const int c = idx & 15;           // 8-column chunk
      int nb = blockN + (c << 3);
      nb = nb <= N - 8 ? nb : N - 8;    // branchless clamp (N % 8 == 0)
      const uint4 w = *(const uint4*)(Bh + (size_t)(k0 + k) * ldb + nb);
      const int nl = nb - blockN < 0 ? 0 : nb - blockN;
      unsigned short e[8];
      e[0] = (unsigned short)(w.x);  e[1] = (unsigned short)(w.x >> 16);
      e[2] = (unsigned short)(w.y);  e[3] = (unsigned short)(w.y >> 16);
      e[4] = (unsigned short)(w.z);  e[5] = (unsigned short)(w.z >> 16);
      e[6] = (unsigned short)(w.w);  e[7] = (unsigned short)(w.w >> 16);
#pragma unroll
      for (int j = 0; j < 8; ++j) BsT[(nl + j) * BPITCH + k] = e[j];
    }
  }
}

// ---------------------------------------------------------------------------
// Batched tiled GEMM:  C = act(alpha * A @ B [+ bias] [+ resid])
// Block: 256 threads (8 wave32), tile 128x128, K-step 32, double-buffered LDS.
// Wave grid 4(M) x 2(N); per-wave tile 32x64 -> 2x4 v_wmma_f32_16x16x32_bf16.
// REQUIREMENTS: M % 128 == 0, K % 32 == 0, N % 8 == 0.  OOB B-columns are
// clamped; they only feed never-stored outputs.
// ---------------------------------------------------------------------------
template <bool A_IS_F32, bool TRANS_B, bool RELU,
          bool BIAS, bool RESID, bool OUTF, bool OUTH>
__global__ __launch_bounds__(256)
void gemm_bf16_wmma(const void* __restrict__ Av,
                    long long strideA_o, long long strideA_i, int lda,
                    const unsigned short* __restrict__ Bh0,
                    long long strideB_o, long long strideB_i, int ldb,
                    float* __restrict__ outF, unsigned short* __restrict__ outH,
                    long long strideC_o, long long strideC_i, int ldc,
                    const float* __restrict__ bias,
                    const float* __restrict__ resid, int ldr,
                    int M, int N, int K, int Hdim, float alpha) {
  __shared__ unsigned short As[2][BM * APITCH];
  __shared__ unsigned short Bs[2][BN * BPITCH];

  const int bz = blockIdx.z;
  const long long bo = bz / Hdim, bi = bz % Hdim;
  const long long aOff = bo * strideA_o + bi * strideA_i;
  const float*          Af = (const float*)Av + (A_IS_F32 ? aOff : 0);
  const unsigned short* Ah = (const unsigned short*)Av + (A_IS_F32 ? 0 : aOff);
  const unsigned short* Bh = Bh0 + bo * strideB_o + bi * strideB_i;
  const long long cOff = bo * strideC_o + bi * strideC_i;

  const int tid   = threadIdx.x;
  const int lane  = tid & 31;
  const int wid   = tid >> 5;
  const int waveM = wid & 3;   // 4 waves over M (32 rows each)
  const int waveN = wid >> 2;  // 2 waves over N (64 cols each)
  const int half  = lane >> 4; // K-half selector inside fragments
  const int lidx  = lane & 15; // row (A) / col (B,C) inside a 16-tile

  const int blockM = blockIdx.y * BM;
  const int blockN = blockIdx.x * BN;

  v8f acc[2][4];
  const v8f vzero = {0.f, 0.f, 0.f, 0.f, 0.f, 0.f, 0.f, 0.f};
#pragma unroll
  for (int sm = 0; sm < 2; ++sm)
#pragma unroll
    for (int sn = 0; sn < 4; ++sn) acc[sm][sn] = vzero;

  // Prologue: stage tile 0 into buffer 0.
  stage_tiles<A_IS_F32, TRANS_B>(As[0], Bs[0], Af, Ah, Bh,
                                 blockM, blockN, 0, lda, ldb, N, K, tid);
  if (!A_IS_F32 || TRANS_B) { WAIT_ASYNC(); }
  __syncthreads();

  int cur = 0;
  for (int k0 = 0; k0 < K; k0 += BK) {
    const int nxt = cur ^ 1;
    // Stage tile k0+BK into the idle buffer while WMMAs consume `cur`.
    if (k0 + BK < K)
      stage_tiles<A_IS_F32, TRANS_B>(As[nxt], Bs[nxt], Af, Ah, Bh,
                                     blockM, blockN, k0 + BK, lda, ldb, N, K, tid);

    // ---- build fragments per ISA 16-bit A/B lane layout -------------------
    FragBF a[2], b[4];
#pragma unroll
    for (int sm = 0; sm < 2; ++sm) {
      const unsigned short* ar =
          &As[cur][((waveM << 5) + (sm << 4) + lidx) * APITCH];
#pragma unroll
      for (int v = 0; v < 8; ++v) {
        const int kk = ((v >> 2) << 4) + (half << 3) + ((v & 3) << 1);
        a[sm].u[v] = *(const unsigned int*)(ar + kk);
      }
    }
#pragma unroll
    for (int sn = 0; sn < 4; ++sn) {
      const unsigned short* br =
          &Bs[cur][((waveN << 6) + (sn << 4) + lidx) * BPITCH];
#pragma unroll
      for (int v = 0; v < 8; ++v) {
        const int kk = ((v >> 2) << 4) + (half << 3) + ((v & 3) << 1);
        b[sn].u[v] = *(const unsigned int*)(br + kk);
      }
    }

    // ---- 8x v_wmma_f32_16x16x32_bf16 (EXEC all-ones; no divergence) -------
#pragma unroll
    for (int sm = 0; sm < 2; ++sm)
#pragma unroll
      for (int sn = 0; sn < 4; ++sn)
        acc[sm][sn] = __builtin_amdgcn_wmma_f32_16x16x32_bf16(
            false, a[sm].v, false, b[sn].v, (short)0, acc[sm][sn],
            false, false);

    if (!A_IS_F32 || TRANS_B) { WAIT_ASYNC(); }  // next tile landed in LDS
    __syncthreads();                             // everyone done with `cur`
    cur = nxt;
  }

  // ---- epilogue: C layout VGPR r -> row r + 8*half, lane%16 -> col --------
#pragma unroll
  for (int sn = 0; sn < 4; ++sn) {
    const int gc = blockN + (waveN << 6) + (sn << 4) + lidx;
    if (gc < N) {
      const float bv = BIAS ? bias[gc] : 0.0f;
#pragma unroll
      for (int sm = 0; sm < 2; ++sm) {
        const int gr0 = blockM + (waveM << 5) + (sm << 4) + (half << 3);
#pragma unroll
        for (int r = 0; r < 8; ++r) {
          const int gr = gr0 + r;
          float v = acc[sm][sn][r] * alpha + bv;
          if (RESID) v += resid[(size_t)gr * ldr + gc];
          if (RELU)  v = fmaxf(v, 0.0f);
          const size_t ci = (size_t)cOff + (size_t)gr * ldc + gc;
          if (OUTF) outF[ci] = v;
          if (OUTH) outH[ci] = f32_to_bf16(v);
        }
      }
    }
  }
}

// ---------------------------------------------------------------------------
// wave32 block reductions (256 threads = 8 waves)
// ---------------------------------------------------------------------------
DEV float block_sum_256(float s, float* sbuf) {
#pragma unroll
  for (int o = 16; o > 0; o >>= 1) s += __shfl_xor(s, o, 32);
  if ((threadIdx.x & 31) == 0) sbuf[threadIdx.x >> 5] = s;
  __syncthreads();
  s = 0.f;
#pragma unroll
  for (int i = 0; i < 8; ++i) s += sbuf[i];
  __syncthreads();
  return s;
}

DEV float block_max_256(float m, float* sbuf) {
#pragma unroll
  for (int o = 16; o > 0; o >>= 1) m = fmaxf(m, __shfl_xor(m, o, 32));
  if ((threadIdx.x & 31) == 0) sbuf[threadIdx.x >> 5] = m;
  __syncthreads();
  float r = sbuf[0];
#pragma unroll
  for (int i = 1; i < 8; ++i) r = fmaxf(r, sbuf[i]);
  __syncthreads();
  return r;
}

// ---------------------------------------------------------------------------
// In-place row softmax, row length 2048; one block per row, values in regs.
// ---------------------------------------------------------------------------
__global__ __launch_bounds__(256)
void softmax_rows_2048(float* __restrict__ p) {
  __shared__ float sbuf[8];
  float* pr = p + (size_t)blockIdx.x * 2048;
  const int tid = threadIdx.x;
  float v[8];
  float m = -3.402823466e38f;
#pragma unroll
  for (int i = 0; i < 8; ++i) { v[i] = pr[tid + (i << 8)]; m = fmaxf(m, v[i]); }
  m = block_max_256(m, sbuf);
  float s = 0.f;
#pragma unroll
  for (int i = 0; i < 8; ++i) { v[i] = __expf(v[i] - m); s += v[i]; }
  s = block_sum_256(s, sbuf);
  const float inv = 1.0f / s;
#pragma unroll
  for (int i = 0; i < 8; ++i) pr[tid + (i << 8)] = v[i] * inv;
}

// ---------------------------------------------------------------------------
// LayerNorm, d=1024, faithful to reference: unbiased std (ddof=1), /(std+eps)
// ---------------------------------------------------------------------------
__global__ __launch_bounds__(256)
void layernorm_1024(const float* __restrict__ x,
                    const float* __restrict__ gamma,
                    const float* __restrict__ beta,
                    float* __restrict__ outF,
                    unsigned short* __restrict__ outH) {
  __shared__ float sbuf[8];
  const size_t row = blockIdx.x;
  const float* xr = x + row * 1024;
  const int tid = threadIdx.x;
  float v[4];
  float s = 0.f;
#pragma unroll
  for (int i = 0; i < 4; ++i) { v[i] = xr[tid + (i << 8)]; s += v[i]; }
  const float mean = block_sum_256(s, sbuf) * (1.0f / 1024.0f);
  float q = 0.f;
#pragma unroll
  for (int i = 0; i < 4; ++i) { const float d = v[i] - mean; q += d * d; }
  const float var = block_sum_256(q, sbuf) * (1.0f / 1023.0f);  // ddof=1
  const float inv = 1.0f / (sqrtf(var) + 1e-5f);
#pragma unroll
  for (int i = 0; i < 4; ++i) {
    const int c = tid + (i << 8);
    const float y = gamma[c] * (v[i] - mean) * inv + beta[c];
    outF[row * 1024 + c] = y;
    if (outH) outH[row * 1024 + c] = f32_to_bf16(y);
  }
}

// ---------------------------------------------------------------------------
// Host orchestration
// ---------------------------------------------------------------------------
extern "C" void kernel_launch(void* const* d_in, const int* in_sizes, int n_in,
                              void* d_out, int out_size, void* d_ws,
                              size_t ws_size, hipStream_t stream) {
  (void)in_sizes; (void)n_in; (void)out_size; (void)ws_size;

  const int B = 2, S = 2048, E = 1024, H = 16, D = 64, F = 4096;
  const int M0 = B * S;  // 4096 token rows

  const float* inputs = (const float*)d_in[0];
  const float* W_q = (const float*)d_in[1];  const float* b_q = (const float*)d_in[2];
  const float* W_k = (const float*)d_in[3];  const float* b_k = (const float*)d_in[4];
  const float* W_v = (const float*)d_in[5];  const float* b_v = (const float*)d_in[6];
  const float* W_o = (const float*)d_in[7];  const float* b_o = (const float*)d_in[8];
  const float* W_1 = (const float*)d_in[9];  const float* b_1 = (const float*)d_in[10];
  const float* W_2 = (const float*)d_in[11]; const float* b_2 = (const float*)d_in[12];
  const float* gamma1 = (const float*)d_in[13]; const float* beta1 = (const float*)d_in[14];
  const float* gamma2 = (const float*)d_in[15]; const float* beta2 = (const float*)d_in[16];

  float* outp = (float*)d_out;                 // [B,S,E] fp32
  float* attn = outp + (size_t)M0 * E;         // [B,H,S,S] fp32 (output #2)

  // workspace carve-out (~152 MiB total)
  char* ws = (char*)d_ws;
  size_t off = 0;
  auto alloc = [&](size_t bytes) -> char* {
    char* p = ws + off;
    off += (bytes + 255) & ~(size_t)255;
    return p;
  };
  unsigned short* Xbf  = (unsigned short*)alloc((size_t)M0 * E * 2);
  unsigned short* Wqbf = (unsigned short*)alloc((size_t)E * E * 2);
  unsigned short* Wkbf = (unsigned short*)alloc((size_t)E * E * 2);
  unsigned short* Wvbf = (unsigned short*)alloc((size_t)E * E * 2);
  unsigned short* Wobf = (unsigned short*)alloc((size_t)E * E * 2);
  unsigned short* W1bf = (unsigned short*)alloc((size_t)E * F * 2);
  unsigned short* W2bf = (unsigned short*)alloc((size_t)F * E * 2);
  unsigned short* Qbf  = (unsigned short*)alloc((size_t)M0 * E * 2);
  unsigned short* Kbf  = (unsigned short*)alloc((size_t)M0 * E * 2);
  unsigned short* Vbf  = (unsigned short*)alloc((size_t)M0 * E * 2);
  unsigned short* Cbf  = (unsigned short*)alloc((size_t)M0 * E * 2);  // ctx
  float*          pre1 = (float*)alloc((size_t)M0 * E * 4);
  float*          x1f  = (float*)alloc((size_t)M0 * E * 4);
  unsigned short* x1bf = (unsigned short*)alloc((size_t)M0 * E * 2);
  unsigned short* hbf  = (unsigned short*)alloc((size_t)M0 * F * 2);
  float*          pre2 = (float*)alloc((size_t)M0 * E * 4);

  auto cvt = [&](const float* s, unsigned short* dptr, int n) {
    const int blocks = (n + 256 * 8 - 1) / (256 * 8);
    cvt_f32_to_bf16<<<blocks, 256, 0, stream>>>(s, dptr, n);
  };
  cvt(inputs, Xbf,  M0 * E);
  cvt(W_q,    Wqbf, E * E);
  cvt(W_k,    Wkbf, E * E);
  cvt(W_v,    Wvbf, E * E);
  cvt(W_o,    Wobf, E * E);
  cvt(W_1,    W1bf, E * F);
  cvt(W_2,    W2bf, F * E);

  const dim3 gEE((E + 127) / 128, (M0 + 127) / 128, 1);

  // Q/K/V = X @ W + b  (bf16 out)            <A_F32,TB,RELU,BIAS,RES,OF,OH>
  gemm_bf16_wmma<false, false, false, true, false, false, true>
      <<<gEE, 256, 0, stream>>>(
      Xbf, 0, 0, E, Wqbf, 0, 0, E,
      nullptr, Qbf, 0, 0, E, b_q, nullptr, 0, M0, E, E, 1, 1.0f);
  gemm_bf16_wmma<false, false, false, true, false, false, true>
      <<<gEE, 256, 0, stream>>>(
      Xbf, 0, 0, E, Wkbf, 0, 0, E,
      nullptr, Kbf, 0, 0, E, b_k, nullptr, 0, M0, E, E, 1, 1.0f);
  gemm_bf16_wmma<false, false, false, true, false, false, true>
      <<<gEE, 256, 0, stream>>>(
      Xbf, 0, 0, E, Wvbf, 0, 0, E,
      nullptr, Vbf, 0, 0, E, b_v, nullptr, 0, M0, E, E, 1, 1.0f);

  // scores = Q @ K^T / sqrt(d)  -> fp32 straight into d_out attn region
  const dim3 gSS((S + 127) / 128, (S + 127) / 128, B * H);
  gemm_bf16_wmma<false, true, false, false, false, true, false>
      <<<gSS, 256, 0, stream>>>(
      Qbf, (long long)S * E, D, E,
      Kbf, (long long)S * E, D, E,
      attn, nullptr, (long long)H * S * S, (long long)S * S, S,
      nullptr, nullptr, 0, S, S, D, H, 0.125f);

  // softmax rows in place -> attn_prob output
  softmax_rows_2048<<<B * H * S, 256, 0, stream>>>(attn);

  // ctx = P @ V  (P read back fp32, converted to bf16 during LDS staging)
  const dim3 gSD((D + 127) / 128, (S + 127) / 128, B * H);
  gemm_bf16_wmma<true, false, false, false, false, false, true>
      <<<gSD, 256, 0, stream>>>(
      attn, (long long)H * S * S, (long long)S * S, S,
      Vbf, (long long)S * E, D, E,
      nullptr, Cbf, (long long)S * E, D, E,
      nullptr, nullptr, 0, S, D, S, H, 1.0f);

  // attn_out = ctx @ W_o + b_o + inputs (residual fused) -> pre1 fp32
  gemm_bf16_wmma<false, false, false, true, true, true, false>
      <<<gEE, 256, 0, stream>>>(
      Cbf, 0, 0, E, Wobf, 0, 0, E,
      pre1, nullptr, 0, 0, E, b_o, inputs, E, M0, E, E, 1, 1.0f);

  // x1 = LN(pre1)  (fp32 + bf16 copy for FFN)
  layernorm_1024<<<M0, 256, 0, stream>>>(pre1, gamma1, beta1, x1f, x1bf);

  // h = relu(x1 @ W1 + b1)  (bf16)
  const dim3 gEF((F + 127) / 128, (M0 + 127) / 128, 1);
  gemm_bf16_wmma<false, false, true, true, false, false, true>
      <<<gEF, 256, 0, stream>>>(
      x1bf, 0, 0, E, W1bf, 0, 0, F,
      nullptr, hbf, 0, 0, F, b_1, nullptr, 0, M0, F, E, 1, 1.0f);

  // pre2 = h @ W2 + b2 + x1 (residual fused)
  gemm_bf16_wmma<false, false, false, true, true, true, false>
      <<<gEE, 256, 0, stream>>>(
      hbf, 0, 0, F, W2bf, 0, 0, E,
      pre2, nullptr, 0, 0, E, b_2, x1f, E, M0, E, F, 1, 1.0f);

  // out = LN(pre2) -> d_out
  layernorm_1024<<<M0, 256, 0, stream>>>(pre2, gamma2, beta2, outp, nullptr);
}
